// Encoder_CN_PI_7164005449983
// MI455X (gfx1250) — matrix operand order repaired and checked by reading, not compiled, not run on hardware
//
#include <hip/hip_runtime.h>
#include <hip/hip_bf16.h>
#include <math.h>

// ---------------- model constants ----------------
#define BB      128           // batch
#define TT      192           // time (== MAX_LEN_PAD)
#define CXC     512           // DIM_ENC
#define CFC     256           // DIM_ENC3
#define NSEG    7             // MAX_NUM_SEG = 128//19 + 1
#define LSEG    64            // 2*MAX_LEN_SEG
#define NJ      (NSEG*LSEG)   // 448

typedef __attribute__((ext_vector_type(16))) __bf16 v16bf;
typedef __attribute__((ext_vector_type(8)))  __bf16 v8bf;
typedef __attribute__((ext_vector_type(8)))  float  v8f;

__device__ __forceinline__ float sigmoidf_(float x){ return 1.0f/(1.0f+expf(-x)); }

// ---------------- weight cast f32 -> bf16, K padded to Kp ----------------
__global__ void __launch_bounds__(256)
castw_kernel(const float* __restrict__ src, __bf16* __restrict__ dst,
             int K, int Kp, int total)
{
    int i = blockIdx.x*256 + threadIdx.x;
    if (i >= total) return;
    int o = i / Kp, k = i - o*Kp;
    dst[i] = (k < K) ? (__bf16)src[(size_t)o*K + k] : (__bf16)0.0f;
}

// ---------------- conv1d(k=5, pad=2) as WMMA GEMM ----------------
// A = weights (O x Kp) bf16 row-major, Kp = Cin_pad*5, Cin_pad = ceil64(Cin).
// Block (256 thr, 8 waves) computes 128(O) x 64(T) for one batch element.
// Input tile (64 ch x 68 t, f32, zero-padded) is staged into LDS with
// gfx1250 async-to-LDS loads; each wave then computes 4 O-tiles x 1 T-tile,
// so one im2col B-fragment feeds 4 v_wmma_f32_16x16x32_bf16 ops.
__global__ void __launch_bounds__(256)
conv_wmma_kernel(const float* __restrict__ in, int in_bstride, int Cin,
                 const __bf16* __restrict__ wbf, int Kp,
                 const float* __restrict__ bias,
                 float* __restrict__ out, int O)
{
    __shared__ float tile[64*68];

    const int wave = threadIdx.x >> 5;
    const int lane = threadIdx.x & 31;
    const int hs   = lane >> 4;                 // half-select
    const int mr   = lane & 15;                 // row (A) / col (B,C)
    const int b    = blockIdx.z;
    const int tB   = blockIdx.y*64;             // block time base
    const int o0   = blockIdx.x*128 + (wave & 1)*64;
    const int t0   = tB + (wave >> 1)*16;
    const int tloc = (t0 - tB) + mr;            // 0..63 inside tile

    const float* xb = in + (size_t)b*in_bstride;
    const int nchunk = Kp / 320;                // channel chunks of 64

    v8f acc[4] = {};

    for (int cch = 0; cch < nchunk; ++cch) {
        const int c0 = cch*64;
        // ---- stage 64ch x 68t input tile into LDS (async where in-range) ----
        for (int it = 0; it < 17; ++it) {            // 64*68/256 == 17 exact
            int idx = threadIdx.x + it*256;
            int ch  = idx / 68, u = idx - ch*68;
            int ci  = c0 + ch, tt = tB - 2 + u;
            unsigned laddr = (unsigned)(uintptr_t)&tile[idx];
            if (ci < Cin && tt >= 0 && tt < TT) {
                const float* g = xb + (size_t)ci*TT + tt;
                asm volatile("global_load_async_to_lds_b32 %0, %1, off"
                             :: "v"(laddr), "v"(g) : "memory");
            } else {
                tile[idx] = 0.0f;
            }
        }
        asm volatile("s_wait_asynccnt 0x0" ::: "memory");
        __syncthreads();

        // ---- 10 sub-chunks of K=32 over this channel chunk ----
        for (int ks = 0; ks < 10; ++ks) {
            const int kcg = c0*5 + ks*32;       // global K base (A side)
            // B fragment from LDS: lanes 0-15 hold K +0..15, lanes 16-31 +16..31
            int ikl = ks*32 + hs*16;
            int cil = ikl / 5;
            int kk  = ikl - cil*5;
            v16bf bm;
#pragma unroll
            for (int e = 0; e < 16; ++e) {
                bm[e] = (__bf16)tile[cil*68 + tloc + kk];
                if (++kk == 5) { kk = 0; ++cil; }
            }
            // 4 O-tiles share this B fragment
#pragma unroll
            for (int oi = 0; oi < 4; ++oi) {
                const __bf16* arow = wbf + (size_t)(o0 + oi*16 + mr)*Kp + hs*16 + kcg;
                v8bf alo = *(const v8bf*)(arow);
                v8bf ahi = *(const v8bf*)(arow + 8);
                v16bf a;
#pragma unroll
                for (int e = 0; e < 8; ++e) { a[e] = alo[e]; a[e+8] = ahi[e]; }
                acc[oi] = __builtin_amdgcn_wmma_f32_16x16x32_bf16(
                              false, a, false, bm, (short)0, acc[oi], false, false);
            }
        }
        __syncthreads();
    }

    // C/D layout: vgpr v, lanes 0-15 -> M=v; lanes 16-31 -> M=8+v; N=lane&15
#pragma unroll
    for (int oi = 0; oi < 4; ++oi) {
        float* op = out + ((size_t)b*O + o0 + oi*16)*TT + (t0 + mr);
#pragma unroll
        for (int v = 0; v < 8; ++v) {
            int m = hs*8 + v;
            op[(size_t)m*TT] = acc[oi][v] + bias[o0 + oi*16 + m];
        }
    }
}

// ---------------- GroupNorm(16 ch/group) + ReLU, in place ----------------
__global__ void __launch_bounds__(256)
gn_relu_kernel(float* __restrict__ y, const float* __restrict__ gamma,
               const float* __restrict__ beta, int C)
{
    const int b = blockIdx.x, g = blockIdx.y;
    const size_t base = ((size_t)b*C + g*16)*TT;
    const int n = 16*TT;
    float s = 0.0f, q = 0.0f;
    for (int i = threadIdx.x; i < n; i += 256) { float v = y[base+i]; s += v; q += v*v; }
    __shared__ float rs[256], rq[256];
    rs[threadIdx.x] = s; rq[threadIdx.x] = q;
    __syncthreads();
    for (int st = 128; st > 0; st >>= 1) {
        if (threadIdx.x < st) { rs[threadIdx.x] += rs[threadIdx.x+st];
                                rq[threadIdx.x] += rq[threadIdx.x+st]; }
        __syncthreads();
    }
    float mu   = rs[0] / n;
    float var  = rq[0] / n - mu*mu;
    float rstd = rsqrtf(var + 1e-5f);
    for (int i = threadIdx.x; i < n; i += 256) {
        int c = g*16 + i/TT;
        float v = (y[base+i] - mu)*rstd*gamma[c] + beta[c];
        y[base+i] = v > 0.0f ? v : 0.0f;
    }
}

// ---------------- interp_lnr (random resampling), one block per batch ----
__global__ void __launch_bounds__(256)
interp_kernel(const float* __restrict__ xs, const float* __restrict__ fs,
              const float* __restrict__ scales, const int* __restrict__ seg,
              float* __restrict__ xd, float* __restrict__ fd)
{
    const int b = blockIdx.x;
    __shared__ int   mpos[NJ];
    __shared__ int   midx[NJ];
    __shared__ float mlam[NJ];
    if (threadIdx.x == 0) {
        int off = 0, pos = 0;
        for (int s = 0; s < NSEG; ++s) {
            float sc = scales[b*NSEG + s] + 0.5f;
            int   ls = seg[b*NSEG + s] + 19;          // + MIN_LEN_SEG
            for (int l = 0; l < LSEG; ++l) {
                int   j   = s*LSEG + l;
                float idx = (float)l / sc;
                float fl  = floorf(idx);
                float lam = idx - fl;
                int   io  = (int)fl + off;
                bool  m   = (fl < (float)(ls - 1)) && (io < TT - 1);
                int   p   = -1;
                if (m) { if (pos < TT) p = pos; pos++; }
                mpos[j] = p;
                int ic = io; if (ic < 0) ic = 0; if (ic > TT-2) ic = TT-2;
                midx[j] = ic;
                mlam[j] = lam;
            }
            off += ls;
        }
    }
    __syncthreads();
    for (int i = threadIdx.x; i < CXC*TT; i += 256) xd[(size_t)b*CXC*TT + i] = 0.0f;
    for (int i = threadIdx.x; i < CFC*TT; i += 256) fd[(size_t)b*CFC*TT + i] = 0.0f;
    __syncthreads();
    for (int j = 0; j < NJ; ++j) {
        int p = mpos[j]; if (p < 0) continue;
        int ic = midx[j]; float lam = mlam[j];
        for (int c = threadIdx.x; c < CXC + CFC; c += 256) {
            if (c < CXC) {
                const float* sp = xs + ((size_t)b*CXC + c)*TT;
                xd[((size_t)b*CXC + c)*TT + p] = (1.0f-lam)*sp[ic] + lam*sp[ic+1];
            } else {
                int cf = c - CXC;
                const float* sp = fs + ((size_t)b*CFC + cf)*TT;
                fd[((size_t)b*CFC + cf)*TT + p] = (1.0f-lam)*sp[ic] + lam*sp[ic+1];
            }
        }
    }
}

// ---------------- LSTM input-gates precompute: gx = x . Wih^T + b --------
__global__ void __launch_bounds__(128)
gx_kernel(const float* __restrict__ in, int sB, int sC, int sT,
          const float* __restrict__ wih, const float* __restrict__ bias,
          int K, float* __restrict__ out)
{
    int idx = blockIdx.x*128 + threadIdx.x;     // B*T*32 total, exact grid
    int g  = idx & 31;
    int bt = idx >> 5;
    int t  = bt % TT, b = bt / TT;
    const float* w = wih + (size_t)g*K;
    const float* x = in  + (size_t)b*sB + (size_t)t*sT;
    float acc = bias[g];
    for (int c = 0; c < K; ++c) acc += w[c]*x[(size_t)c*sC];
    out[idx] = acc;
}

// ---------------- LSTM scan (H=8), one wave per (batch, direction) ------
__global__ void __launch_bounds__(32)
lstm_scan_kernel(const float* __restrict__ gxf, const float* __restrict__ gxb,
                 const float* __restrict__ whh, float* __restrict__ out)
{
    const int b = blockIdx.x, d = blockIdx.y;
    const float* gx = d ? gxb : gxf;
    const float* W  = whh + d*32*8;
    __shared__ float h[8], cst[8], ga[32];
    int lane = threadIdx.x;
    if (lane < 8) { h[lane] = 0.0f; cst[lane] = 0.0f; }
    __syncthreads();
    for (int st = 0; st < TT; ++st) {
        int t = d ? (TT-1-st) : st;
        float g = gx[((size_t)b*TT + t)*32 + lane];
        const float* wr = W + lane*8;
#pragma unroll
        for (int j = 0; j < 8; ++j) g += h[j]*wr[j];
        ga[lane] = g;
        __syncthreads();
        if (lane < 8) {
            float iv = sigmoidf_(ga[lane]);
            float fv = sigmoidf_(ga[8+lane]);
            float gv = tanhf(ga[16+lane]);
            float ov = sigmoidf_(ga[24+lane]);
            float c2 = fv*cst[lane] + iv*gv;
            cst[lane] = c2;
            float hv = ov*tanhf(c2);
            h[lane] = hv;
            out[((size_t)b*TT + t)*16 + d*8 + lane] = hv;
        }
        __syncthreads();
    }
}

// ---------------- downsample codes: concat(fw[7::8], bw[::8]) -----------
__global__ void __launch_bounds__(256)
codes_kernel(const float* __restrict__ h, float* __restrict__ out)
{
    int idx = blockIdx.x*256 + threadIdx.x;     // B*24*16
    if (idx >= BB*24*16) return;
    int c = idx & 15;
    int s = (idx >> 4) % 24;
    int b = idx / (24*16);
    int t = (c < 8) ? (s*8 + 7) : (s*8);
    out[idx] = h[((size_t)b*TT + t)*16 + c];
}

// ---------------- embedding: out = flat . W^T + b (K=384) ---------------
__global__ void __launch_bounds__(128)
emb_kernel(const float* __restrict__ flat, const float* __restrict__ W,
           const float* __restrict__ bias, float* __restrict__ out, int N)
{
    int idx = blockIdx.x*128 + threadIdx.x;
    if (idx >= BB*N) return;
    int e = idx % N, b = idx / N;
    const float* f = flat + (size_t)b*384;
    const float* w = W + (size_t)e*384;
    float acc = bias[e];
    for (int k = 0; k < 384; ++k) acc += f[k]*w[k];
    out[idx] = acc;
}

// ======================= host orchestration =============================
extern "C" void kernel_launch(void* const* d_in, const int* in_sizes, int n_in,
                              void* d_out, int out_size, void* d_ws, size_t ws_size,
                              hipStream_t stream)
{
    (void)in_sizes; (void)n_in; (void)out_size; (void)ws_size;
    const float* x_f0    = (const float*)d_in[0];
    const float* scales  = (const float*)d_in[1];
    const int*   segd    = (const int*)  d_in[2];
    const float* c1w[3]  = {(const float*)d_in[3],(const float*)d_in[4],(const float*)d_in[5]};
    const float* c1b     = (const float*)d_in[6];
    const float* g1g     = (const float*)d_in[7];
    const float* g1b     = (const float*)d_in[8];
    const float* c2w[3]  = {(const float*)d_in[9],(const float*)d_in[10],(const float*)d_in[11]};
    const float* c2b     = (const float*)d_in[12];
    const float* g2g     = (const float*)d_in[13];
    const float* g2b     = (const float*)d_in[14];
    const float* l1_wih0 = (const float*)d_in[15];
    const float* l1_whh0 = (const float*)d_in[16];
    const float* l1_b0   = (const float*)d_in[17];
    const float* l1_wih1 = (const float*)d_in[18];
    const float* l1_whh1 = (const float*)d_in[19];
    const float* l1_b1   = (const float*)d_in[20];
    const float* l2_wih  = (const float*)d_in[21];
    const float* l2_whh  = (const float*)d_in[22];
    const float* l2_b    = (const float*)d_in[23];
    const float* e1w     = (const float*)d_in[24];
    const float* e1b     = (const float*)d_in[25];
    const float* e2w     = (const float*)d_in[26];
    const float* e2b     = (const float*)d_in[27];
    float* out = (float*)d_out;

    // ---- workspace layout (all 16B aligned) ----
    float* ws = (float*)d_ws;
    const size_t XB = (size_t)BB*CXC*TT;
    const size_t FB = (size_t)BB*CFC*TT;
    float* xbuf0 = ws;
    float* xbuf1 = xbuf0 + XB;
    float* fbuf0 = xbuf1 + XB;
    float* fbuf1 = fbuf0 + FB;
    float* gxf   = fbuf1 + FB;
    float* gxb   = gxf + (size_t)BB*TT*32;
    float* hx0   = gxb + (size_t)BB*TT*32;
    float* hx1   = hx0 + (size_t)BB*TT*16;
    float* hf    = hx1 + (size_t)BB*TT*16;
    __bf16* wb   = (__bf16*)(hf + (size_t)BB*TT*16);

    // padded K: Kp = ceil64(Cin)*5  (multiple of 320, so also of 32)
    const int Kx0 = 400,  Kpx0 = 640;    // Cin  80 -> 128
    const int Kx  = 2560, Kpx  = 2560;   // Cin 512 -> 512
    const int Kf0 = 1285, Kpf0 = 1600;   // Cin 257 -> 320
    const int Kf  = 1280, Kpf  = 1280;   // Cin 256 -> 256
    __bf16* wx[3]; __bf16* wf[3];
    size_t woff = 0;
    wx[0] = wb + woff; woff += (size_t)512*Kpx0;
    wx[1] = wb + woff; woff += (size_t)512*Kpx;
    wx[2] = wb + woff; woff += (size_t)512*Kpx;
    wf[0] = wb + woff; woff += (size_t)256*Kpf0;
    wf[1] = wb + woff; woff += (size_t)256*Kpf;
    wf[2] = wb + woff;

    // ---- cast conv weights to bf16 (zero-padded K) ----
    castw_kernel<<<(512*Kpx0+255)/256,256,0,stream>>>(c1w[0], wx[0], Kx0, Kpx0, 512*Kpx0);
    castw_kernel<<<(512*Kpx +255)/256,256,0,stream>>>(c1w[1], wx[1], Kx , Kpx , 512*Kpx );
    castw_kernel<<<(512*Kpx +255)/256,256,0,stream>>>(c1w[2], wx[2], Kx , Kpx , 512*Kpx );
    castw_kernel<<<(256*Kpf0+255)/256,256,0,stream>>>(c2w[0], wf[0], Kf0, Kpf0, 256*Kpf0);
    castw_kernel<<<(256*Kpf +255)/256,256,0,stream>>>(c2w[1], wf[1], Kf , Kpf , 256*Kpf );
    castw_kernel<<<(256*Kpf +255)/256,256,0,stream>>>(c2w[2], wf[2], Kf , Kpf , 256*Kpf );

    // ---- 3 x (conv + GN/ReLU for both paths, then joint resample) ----
    for (int i = 0; i < 3; ++i) {
        const float* xin = (i == 0) ? x_f0 : xbuf0;
        int xbs  = (i == 0) ? 337*TT : CXC*TT;
        int xcin = (i == 0) ? 80 : 512;
        int xkp  = (i == 0) ? Kpx0 : Kpx;
        conv_wmma_kernel<<<dim3(4,3,BB),256,0,stream>>>(xin, xbs, xcin, wx[i], xkp,
                                                        c1b + i*512, xbuf1, 512);
        gn_relu_kernel<<<dim3(BB,32),256,0,stream>>>(xbuf1, g1g + i*512, g1b + i*512, 512);

        const float* fin = (i == 0) ? (x_f0 + 80*TT) : fbuf0;
        int fbs  = (i == 0) ? 337*TT : CFC*TT;
        int fcin = (i == 0) ? 257 : 256;
        int fkp  = (i == 0) ? Kpf0 : Kpf;
        conv_wmma_kernel<<<dim3(2,3,BB),256,0,stream>>>(fin, fbs, fcin, wf[i], fkp,
                                                        c2b + i*256, fbuf1, 256);
        gn_relu_kernel<<<dim3(BB,16),256,0,stream>>>(fbuf1, g2g + i*256, g2b + i*256, 256);

        interp_kernel<<<BB,256,0,stream>>>(xbuf1, fbuf1,
                                           scales + i*BB*NSEG, segd + i*BB*NSEG,
                                           xbuf0, fbuf0);
    }

    // ---- bi-LSTMs ----
    const int ngx = BB*TT*32;   // exact multiple of 128
    gx_kernel<<<ngx/128,128,0,stream>>>(xbuf0, CXC*TT, TT, 1, l1_wih0,        l1_b0,      512, gxf);
    gx_kernel<<<ngx/128,128,0,stream>>>(xbuf0, CXC*TT, TT, 1, l1_wih0+32*512, l1_b0 + 32, 512, gxb);
    lstm_scan_kernel<<<dim3(BB,2),32,0,stream>>>(gxf, gxb, l1_whh0, hx0);
    gx_kernel<<<ngx/128,128,0,stream>>>(hx0, TT*16, 1, 16, l1_wih1,        l1_b1,      16, gxf);
    gx_kernel<<<ngx/128,128,0,stream>>>(hx0, TT*16, 1, 16, l1_wih1+32*16,  l1_b1 + 32, 16, gxb);
    lstm_scan_kernel<<<dim3(BB,2),32,0,stream>>>(gxf, gxb, l1_whh1, hx1);
    gx_kernel<<<ngx/128,128,0,stream>>>(fbuf0, CFC*TT, TT, 1, l2_wih,        l2_b,      256, gxf);
    gx_kernel<<<ngx/128,128,0,stream>>>(fbuf0, CFC*TT, TT, 1, l2_wih+32*256, l2_b + 32, 256, gxb);
    lstm_scan_kernel<<<dim3(BB,2),32,0,stream>>>(gxf, gxb, l2_whh, hf);

    // ---- codes + embeddings (straight into d_out, reference order) ----
    // layout: codes_x[0,49152) emb1[49152,81920) codes_f0[81920,131072) emb2[131072,147456)
    codes_kernel<<<(BB*24*16)/256,256,0,stream>>>(hx1, out);
    codes_kernel<<<(BB*24*16)/256,256,0,stream>>>(hf,  out + 81920);
    emb_kernel<<<(BB*256)/128,128,0,stream>>>(out,          e1w, e1b, out + 49152, 256);
    emb_kernel<<<(BB*128)/128,128,0,stream>>>(out + 81920,  e2w, e2b, out + 131072, 128);
}
